// ManyBodyVoxel_26680336843020
// MI455X (gfx1250) — matrix operand order repaired and checked by reading, not compiled, not run on hardware
//
#include <hip/hip_runtime.h>

typedef __attribute__((ext_vector_type(16))) _Float16 v16h;
typedef __attribute__((ext_vector_type(8)))  _Float16 v8h;
typedef __attribute__((ext_vector_type(8)))  float    v8f;

#define GRID   16
#define NATOM  64
#define NCH    10

static __device__ __forceinline__ v16h cat8(v8h lo, v8h hi) {
    return __builtin_shufflevector(lo, hi, 0,1,2,3,4,5,6,7,8,9,10,11,12,13,14,15);
}

// One block per (a, channel). 128 threads = 4 waves (wave32).
// box[x][y][z] = sum_n Ex[n][x] * Ey[n][y] * Ez[n][z]  ==  (16x64) @ (64x256) GEMM on WMMA.
__global__ __launch_bounds__(128)
void manybody_voxel_wmma(const float* __restrict__ dvec,   // (1,128,64,3) f32
                         const float* __restrict__ sigma,  // (1,) f32
                         const int*   __restrict__ zed,    // (128,64) int
                         float* __restrict__ out,          // (1,128,10,16,16,16) f32
                         int nch_blocks)
{
    // A-matrix staged [M=x][K=n] so each lane's fragment chunks are contiguous.
    __shared__ alignas(16) _Float16 ExT[GRID * NATOM];    // [x][n]   2 KB
    __shared__ alignas(16) _Float16 Ey [NATOM * GRID];    // [n][y]   2 KB
    __shared__ alignas(16) _Float16 Ez [NATOM * GRID];    // [n][z]   2 KB
    __shared__ alignas(16) _Float16 BT [256 * NATOM];     // [m][n]  32 KB, m=y*16+z

    const int tid = threadIdx.x;
    const int bid = blockIdx.x;
    if (bid >= nch_blocks) return;
    const int a  = bid / NCH;
    const int c  = bid % NCH;
    const int zi = c >> 1;                       // atom-type index 0..4
    const int Zt = (zi == 0) ? 1 : (zi == 1) ? 6 : (zi == 2) ? 7 : (zi == 3) ? 8 : 16;
    const float L  = (c & 1) ? 12.0f : 8.0f;
    const float s  = sigma[0];
    const float coeff = -0.5f / (s * s);
    const float t0 = -0.5f * L;
    const float dt = L * (1.0f / 15.0f);

    // ---- Phase 1: separable exp rows per (axis, atom): 192 tasks x 16 exps ----
    for (int task = tid; task < 3 * NATOM; task += 128) {
        const int axis = task >> 6;              // 0,1,2
        const int n    = task & (NATOM - 1);
        const float msk = (zed[a * NATOM + n] == Zt) ? 1.0f : 0.0f;
        const float d   = dvec[(a * NATOM + n) * 3 + axis] * msk;
        #pragma unroll
        for (int i = 0; i < GRID; ++i) {
            const float u = t0 + dt * (float)i - d;
            const _Float16 e = (_Float16)__expf(coeff * u * u);
            if (axis == 0)      ExT[i * NATOM + n] = e;   // A^T staging
            else if (axis == 1) Ey[n * GRID + i]   = e;
            else                Ez[n * GRID + i]   = e;
        }
    }
    __syncthreads();

    // ---- Phase 2: B^T[m][n] = Ey[n][y] * Ez[n][z], m = y*16+z (fragment-contiguous) ----
    for (int idx = tid; idx < 256 * NATOM; idx += 128) {
        const int m = idx >> 6;                  // 0..255
        const int n = idx & (NATOM - 1);
        BT[idx] = Ey[n * GRID + (m >> 4)] * Ez[n * GRID + (m & 15)];
    }
    __syncthreads();

    // ---- Phase 3: 16x256x64 GEMM via v_wmma_f32_16x16x32_f16 ----
    const int wave = tid >> 5;
    const int lane = tid & 31;
    const int mrow = lane & 15;                  // A: M row / B: N col / D: N col
    const int kgrp = lane >> 4;                  // lane half-group

    // A fragments (shared across all N tiles): 16-bit A 16x32 layout.
    // lane group 0: K=0..15, group 1: K=16..31 (contiguous, matching B).
    const v8h a00 = *(const v8h*)&ExT[mrow * NATOM +  0 + kgrp * 16];
    const v8h a01 = *(const v8h*)&ExT[mrow * NATOM +  8 + kgrp * 16];
    const v8h a10 = *(const v8h*)&ExT[mrow * NATOM + 32 + kgrp * 16];
    const v8h a11 = *(const v8h*)&ExT[mrow * NATOM + 40 + kgrp * 16];
    const v16h A0 = cat8(a00, a01);              // K = 0..31
    const v16h A1 = cat8(a10, a11);              // K = 32..63

    float* __restrict__ outp = out + ((size_t)(a * NCH + c) << 12);

    #pragma unroll
    for (int tt = 0; tt < 4; ++tt) {
        const int t   = wave * 4 + tt;           // N tile == y plane
        const int col = t * 16 + mrow;           // B column (m index)
        // B 32x16 in contiguous k order (same as A): lanes 0-15 hold K 0..15, lanes 16-31 hold K 16..31.
        const v8h b00 = *(const v8h*)&BT[col * NATOM + kgrp * 16 + 0];
        const v8h b01 = *(const v8h*)&BT[col * NATOM + kgrp * 16 + 8];
        const v8h b10 = *(const v8h*)&BT[col * NATOM + 32 + kgrp * 16 + 0];
        const v8h b11 = *(const v8h*)&BT[col * NATOM + 32 + kgrp * 16 + 8];
        const v16h B0 = cat8(b00, b01);          // K = 0..31
        const v16h B1 = cat8(b10, b11);          // K = 32..63

        v8f acc = {};
        acc = __builtin_amdgcn_wmma_f32_16x16x32_f16(false, A0, false, B0,
                                                     (short)0, acc, false, false);
        acc = __builtin_amdgcn_wmma_f32_16x16x32_f16(false, A1, false, B1,
                                                     (short)0, acc, false, false);

        // D layout: lane N = lane%16; VGPR r -> M = r + 8*(lane/16). x=M, y=t, z=N.
        #pragma unroll
        for (int r = 0; r < 8; ++r) {
            const int x = r + kgrp * 8;
            outp[x * 256 + t * 16 + mrow] = acc[r];
        }
    }
}

extern "C" void kernel_launch(void* const* d_in, const int* in_sizes, int n_in,
                              void* d_out, int out_size, void* d_ws, size_t ws_size,
                              hipStream_t stream) {
    // setup_inputs order: distance_vector (f32), sigma (f32 scalar), atomic_numbers (int)
    const float* dvec  = (const float*)d_in[0];
    const float* sigma = (const float*)d_in[1];
    const int*   zed   = (const int*)d_in[2];
    float* out = (float*)d_out;

    const int nblocks = out_size >> 12;          // one block per 16^3 box => (a,c) pair
    manybody_voxel_wmma<<<dim3(nblocks), dim3(128), 0, stream>>>(dvec, sigma, zed, out, nblocks);
}
